// ImgtoClass_Metric_25898652795644
// MI455X (gfx1250) — matrix operand order repaired and checked by reading, not compiled, not run on hardware
//
#include <hip/hip_runtime.h>

typedef __attribute__((ext_vector_type(16))) __bf16 v16bf;
typedef __attribute__((ext_vector_type(8)))  float  v8f;

namespace {
constexpr int kB  = 16;    // query batch
constexpr int kC  = 64;    // channels (K dim)
constexpr int kHW = 441;   // 21*21 query descriptors
constexpr int kW  = 16;    // support classes
constexpr int kM  = 2205;  // support descriptors per class
constexpr int kPT = 28;            // ceil(441/16) p-tiles
constexpr int kPPAD = kPT * 16;    // 448
constexpr int kMT = 138;           // ceil(2205/16) m-tiles
constexpr int kKC = 2;             // 64 / 32 k-chunks per WMMA chain
constexpr int kPOT = kPT / 2;      // 14: each wave owns two p-tiles

// workspace byte offsets (all 32B-aligned)
constexpr size_t OFF_A2S  = 0;                                                 // float[kW*kM]
constexpr size_t OFF_A1   = OFF_A2S + sizeof(float) * (size_t)kW * kM;         // float[kB*kPPAD]
constexpr size_t OFF_AS   = OFF_A1  + sizeof(float) * (size_t)kB * kPPAD;      // bf16 frags, support (pre-scaled)
constexpr size_t OFF_BQ   = OFF_AS  + 2ull * kW * kMT * kKC * 512;             // bf16 frags, query
constexpr size_t OFF_PART = OFF_BQ  + 2ull * kB * kPT * kKC * 512;             // float[kB*kW*kPOT]
} // namespace

__device__ __forceinline__ unsigned short f2bf(float f) {
  unsigned int u = __float_as_uint(f);
  u += 0x7FFFu + ((u >> 16) & 1u);   // round-to-nearest-even
  return (unsigned short)(u >> 16);
}

__device__ __forceinline__ void ins3(float c, float& t0, float& t1, float& t2) {
  float mx = fmaxf(t0, c), mn = fminf(t0, c);
  t0 = mx;
  float mx1 = fmaxf(t1, mn), mn1 = fminf(t1, mn);
  t1 = mx1;
  t2 = fmaxf(t2, mn1);
}

// a2s = mask2/||s||  (per support descriptor; folded into the bf16 A fragments)
__global__ void k_sup_scale(const float* __restrict__ x2,
                            const float* __restrict__ mask2,
                            float* __restrict__ a2s) {
  int idx = blockIdx.x * blockDim.x + threadIdx.x;
  if (idx >= kW * kM) return;
  int w = idx / kM, m = idx % kM;
  const float* p = x2 + (size_t)w * kC * kM + m;
  float s = 0.f;
#pragma unroll 8
  for (int c = 0; c < kC; ++c) { float v = p[(size_t)c * kM]; s = fmaf(v, v, s); }
  a2s[idx] = mask2[idx] / sqrtf(s);
}

// a1 = mask1/||q|| (0 on padded p)
__global__ void k_q_scale(const float* __restrict__ x1,
                          const float* __restrict__ mask1,
                          float* __restrict__ a1) {
  int idx = blockIdx.x * blockDim.x + threadIdx.x;
  if (idx >= kB * kPPAD) return;
  int b = idx / kPPAD, p = idx % kPPAD;
  float r = 0.f;
  if (p < kHW) {
    const float* q = x1 + (size_t)b * kC * kHW + p;
    float s = 0.f;
#pragma unroll 8
    for (int c = 0; c < kC; ++c) { float v = q[(size_t)c * kHW]; s = fmaf(v, v, s); }
    r = mask1[(size_t)b * kHW + p] / sqrtf(s);
  }
  a1[idx] = r;
}

// Pack pre-scaled support into 16-bit WMMA A-fragment order:
// lane holds row m = 16*mt + (lane&15); element e -> K = kc*32 + e + (e&8) + 8*(lane>>4)
__global__ void k_packA(const float* __restrict__ x2,
                        const float* __restrict__ a2s,
                        unsigned short* __restrict__ As) {
  int idx = blockIdx.x * blockDim.x + threadIdx.x;       // ((w*MT+mt)*KC+kc)*32+lane
  if (idx >= kW * kMT * kKC * 32) return;
  int lane = idx & 31;
  int kc   = (idx >> 5) & 1;
  int mt   = (idx >> 6) % kMT;
  int w    = (idx >> 6) / kMT;
  int hi = lane >> 4;
  int m  = mt * 16 + (lane & 15);
  float scale = (m < kM) ? a2s[(size_t)w * kM + m] : 0.f;
  unsigned short* dst = As + (size_t)idx * 16;
#pragma unroll
  for (int e = 0; e < 16; ++e) {
    int c = kc * 32 + e + (e & 8) + hi * 8;
    float v = (m < kM) ? x2[((size_t)w * kC + c) * kM + m] * scale : 0.f;
    dst[e] = f2bf(v);
  }
}

// Pack query into 16-bit WMMA B-fragment order:
// lane holds col p = 16*pt + (lane&15); element e -> K = kc*32 + e + (e&8) + 8*(lane>>4)
__global__ void k_packB(const float* __restrict__ x1, unsigned short* __restrict__ Bq) {
  int idx = blockIdx.x * blockDim.x + threadIdx.x;       // ((b*PT+pt)*KC+kc)*32+lane
  if (idx >= kB * kPT * kKC * 32) return;
  int lane = idx & 31;
  int kc   = (idx >> 5) & 1;
  int pt   = (idx >> 6) % kPT;
  int b    = (idx >> 6) / kPT;
  int hi = lane >> 4;
  int p  = pt * 16 + (lane & 15);
  unsigned short* dst = Bq + (size_t)idx * 16;
#pragma unroll
  for (int e = 0; e < 16; ++e) {
    int c = kc * 32 + e + (e & 8) + hi * 8;
    float v = (p < kHW) ? x1[((size_t)b * kC + c) * kHW + p] : 0.f;
    dst[e] = f2bf(v);
  }
}

// One wave per (b, w, pair-of-p-tiles): stream 138 m-tiles through bf16 WMMA.
// A fragments are pre-scaled by mask2/||s||, so the WMMA output IS the masked
// similarity; running top-3 per lane, pad bias only on the peeled last tile.
__global__ void __launch_bounds__(32) k_main(const unsigned short* __restrict__ As,
                                             const unsigned short* __restrict__ Bq,
                                             const float* __restrict__ a1,
                                             float* __restrict__ part) {
  int blk = blockIdx.x;                 // ((b*kW + w)*kPOT + po)
  int po = blk % kPOT;
  int w  = (blk / kPOT) % kW;
  int b  = blk / (kPOT * kW);
  int pt0 = po * 2, pt1 = po * 2 + 1;
  int lane = threadIdx.x;
  int hi = lane >> 4;

  // loop-invariant query B fragments for both p-tiles (register resident)
  const unsigned short* bb0 = Bq + ((size_t)(b * kPT + pt0) * kKC) * 512 + (size_t)lane * 16;
  const unsigned short* bb1 = Bq + ((size_t)(b * kPT + pt1) * kKC) * 512 + (size_t)lane * 16;
  v16bf bA0 = *(const v16bf*)(bb0);
  v16bf bA1 = *(const v16bf*)(bb0 + 512);
  v16bf bB0 = *(const v16bf*)(bb1);
  v16bf bB1 = *(const v16bf*)(bb1 + 512);

  const unsigned short* abase = As + ((size_t)(w * kMT) * kKC) * 512 + (size_t)lane * 16;

  float s0 = -3e38f, s1 = -3e38f, s2 = -3e38f;   // top-3, p-tile 0
  float u0 = -3e38f, u1 = -3e38f, u2 = -3e38f;   // top-3, p-tile 1

  for (int mt = 0; mt < kMT - 1; ++mt) {
    v16bf af0 = *(const v16bf*)(abase + (size_t)mt * 1024);
    v16bf af1 = *(const v16bf*)(abase + (size_t)mt * 1024 + 512);
    v8f acc0 = {}, acc1 = {};
    acc0 = __builtin_amdgcn_wmma_f32_16x16x32_bf16(false, af0, false, bA0, (short)0, acc0, false, false);
    acc1 = __builtin_amdgcn_wmma_f32_16x16x32_bf16(false, af0, false, bB0, (short)0, acc1, false, false);
    acc0 = __builtin_amdgcn_wmma_f32_16x16x32_bf16(false, af1, false, bA1, (short)0, acc0, false, false);
    acc1 = __builtin_amdgcn_wmma_f32_16x16x32_bf16(false, af1, false, bB1, (short)0, acc1, false, false);
#pragma unroll
    for (int r = 0; r < 8; ++r) {
      ins3(acc0[r], s0, s1, s2);
      ins3(acc1[r], u0, u1, u2);
    }
  }
  { // peeled last tile: m = 2192 + 8*hi + r; pad (m >= 2205) -> force -1e30
    constexpr int mt = kMT - 1;
    v16bf af0 = *(const v16bf*)(abase + (size_t)mt * 1024);
    v16bf af1 = *(const v16bf*)(abase + (size_t)mt * 1024 + 512);
    v8f acc0 = {}, acc1 = {};
    acc0 = __builtin_amdgcn_wmma_f32_16x16x32_bf16(false, af0, false, bA0, (short)0, acc0, false, false);
    acc1 = __builtin_amdgcn_wmma_f32_16x16x32_bf16(false, af0, false, bB0, (short)0, acc1, false, false);
    acc0 = __builtin_amdgcn_wmma_f32_16x16x32_bf16(false, af1, false, bA1, (short)0, acc0, false, false);
    acc1 = __builtin_amdgcn_wmma_f32_16x16x32_bf16(false, af1, false, bB1, (short)0, acc1, false, false);
#pragma unroll
    for (int r = 0; r < 8; ++r) {
      float bias = (8 * hi + r >= kM - (kMT - 1) * 16) ? -1e30f : 0.f;
      ins3(acc0[r] + bias, s0, s1, s2);
      ins3(acc1[r] + bias, u0, u1, u2);
    }
  }

  // merge the two half-wave m-partitions (same p lives in lane and lane^16)
  float o0 = __shfl_xor(s0, 16, 32), o1 = __shfl_xor(s1, 16, 32), o2 = __shfl_xor(s2, 16, 32);
  ins3(o0, s0, s1, s2); ins3(o1, s0, s1, s2); ins3(o2, s0, s1, s2);
  float q0 = __shfl_xor(u0, 16, 32), q1 = __shfl_xor(u1, 16, 32), q2 = __shfl_xor(u2, 16, 32);
  ins3(q0, u0, u1, u2); ins3(q1, u0, u1, u2); ins3(q2, u0, u1, u2);

  float val = 0.f;
  if (lane < 16) {
    const float* a1b = a1 + (size_t)b * kPPAD;
    val = (s0 + s1 + s2) * a1b[pt0 * 16 + lane] +
          (u0 + u1 + u2) * a1b[pt1 * 16 + lane];
  }
  for (int off = 16; off; off >>= 1) val += __shfl_down(val, off, 32);
  if (lane == 0) part[blk] = val;
}

__global__ void k_final(const float* __restrict__ part, float* __restrict__ out) {
  int idx = blockIdx.x * blockDim.x + threadIdx.x;   // b*kW + w
  if (idx >= kB * kW) return;
  const float* p = part + (size_t)idx * kPOT;
  float s = 0.f;
#pragma unroll
  for (int i = 0; i < kPOT; ++i) s += p[i];
  out[idx] = s;
}

extern "C" void kernel_launch(void* const* d_in, const int* in_sizes, int n_in,
                              void* d_out, int out_size, void* d_ws, size_t ws_size,
                              hipStream_t stream) {
  (void)in_sizes; (void)n_in; (void)out_size; (void)ws_size;
  const float* x1    = (const float*)d_in[0];
  const float* x2    = (const float*)d_in[1];
  const float* mask1 = (const float*)d_in[2];
  const float* mask2 = (const float*)d_in[3];
  float* out = (float*)d_out;

  char* ws = (char*)d_ws;
  float*          a2s  = (float*)(ws + OFF_A2S);
  float*          a1   = (float*)(ws + OFF_A1);
  unsigned short* As   = (unsigned short*)(ws + OFF_AS);
  unsigned short* Bq   = (unsigned short*)(ws + OFF_BQ);
  float*          part = (float*)(ws + OFF_PART);

  k_sup_scale<<<(kW * kM + 255) / 256, 256, 0, stream>>>(x2, mask2, a2s);
  k_q_scale  <<<(kB * kPPAD + 255) / 256, 256, 0, stream>>>(x1, mask1, a1);
  k_packA    <<<(kW * kMT * kKC * 32 + 255) / 256, 256, 0, stream>>>(x2, a2s, As);
  k_packB    <<<(kB * kPT * kKC * 32 + 255) / 256, 256, 0, stream>>>(x1, Bq);
  k_main     <<<kB * kW * kPOT, 32, 0, stream>>>(As, Bq, a1, part);
  k_final    <<<1, 256, 0, stream>>>(part, out);
}
